// TransformerBlock_41772851921125
// MI455X (gfx1250) — matrix operand order repaired and checked by reading, not compiled, not run on hardware
//
#include <hip/hip_runtime.h>
#include <math.h>

// ---------------- problem constants ----------------
#define BB   16
#define NN   577
#define CC   768
#define HH   12
#define DD   64
#define HID  3072
#define MROWS (BB * NN)          // 9232
#define MPAD  9280               // 145 * 64 (padded row count for bf16 activations)
#define QKV_N (3 * CC)           // 2304
#define NTILES 37                // ceil(577/16)
#define SCOLS  (NTILES * 16)     // 592
#define SPAD   608               // 19*32, K-pad for ctx WMMA

typedef __attribute__((ext_vector_type(16))) __bf16   v16bf;
typedef __attribute__((ext_vector_type(8)))  float    v8f;
typedef __attribute__((ext_vector_type(8)))  unsigned int v8u;
typedef __attribute__((ext_vector_type(4)))  unsigned int u32x4;
typedef __attribute__((ext_vector_type(4)))  int       v4i;

#define AS1 __attribute__((address_space(1)))
#define AS3 __attribute__((address_space(3)))

#if defined(__HIP_DEVICE_COMPILE__) && defined(__gfx1250__) && \
    __has_builtin(__builtin_amdgcn_global_load_async_to_lds_b128)
#define HAS_ASYNC_LDS 1
#else
#define HAS_ASYNC_LDS 0
#endif

// ---------------- helpers ----------------
__device__ __forceinline__ unsigned short f32_to_bf16(float f) {
    unsigned int u = __builtin_bit_cast(unsigned int, f);
    u += 0x7FFFu + ((u >> 16) & 1u);      // round-to-nearest-even
    return (unsigned short)(u >> 16);
}
__device__ __forceinline__ int kA_idx(int i, int hi) {  // A-matrix K, contiguous order (k = 16*hi + s)
    return 2 * i + hi * 16;
}
__device__ __forceinline__ v16bf frag_cast(const unsigned int* u) {
    v8u t;
#pragma unroll
    for (int i = 0; i < 8; ++i) t[i] = u[i];
    return __builtin_bit_cast(v16bf, t);
}
__device__ __forceinline__ v8f wmma_bf16(const unsigned int a[8],
                                         const unsigned int b[8], v8f c) {
    return __builtin_amdgcn_wmma_f32_16x16x32_bf16(
        false, frag_cast(a), false, frag_cast(b), (short)0, c, false, false);
}
__device__ __forceinline__ void wait_ds0() {
    asm volatile("s_wait_dscnt 0" ::: "memory");
}

#if HAS_ASYNC_LDS
__device__ __forceinline__ void async_cp16(const void* g, void* l) {
    // generic->as1 / generic->as3 via integer round-trip (LDS offset = addr[31:0]);
    // builtin wants non-const int4 pointers in addrspace(1)/(3).
    AS1 v4i* gp = (AS1 v4i*)(unsigned long long)g;
    AS3 v4i* lp = (AS3 v4i*)(unsigned int)(unsigned long long)l;
    __builtin_amdgcn_global_load_async_to_lds_b128(gp, lp, 0, 0);
}
__device__ __forceinline__ void async_wait0() {
#if __has_builtin(__builtin_amdgcn_s_wait_asynccnt)
    __builtin_amdgcn_s_wait_asynccnt(0);
#else
    asm volatile("s_wait_asynccnt 0" ::: "memory");
#endif
}
#endif

// ---------------- utility kernels ----------------
__global__ __launch_bounds__(256) void cvt_bf16_kernel(
    const float* __restrict__ in, unsigned short* __restrict__ out, int n) {
    int i = blockIdx.x * 256 + threadIdx.x;
    if (i < n) out[i] = f32_to_bf16(in[i]);
}
__global__ __launch_bounds__(256) void zero_u16_kernel(
    unsigned short* __restrict__ p, int n) {
    int i = blockIdx.x * 256 + threadIdx.x;
    if (i < n) p[i] = 0;
}

// ---------------- layernorm (fp32 in, bf16 out) ----------------
__global__ __launch_bounds__(256) void layernorm_bf16_kernel(
    const float* __restrict__ x, const float* __restrict__ g,
    const float* __restrict__ be, unsigned short* __restrict__ out) {
    __shared__ float red[256];
    const int row = blockIdx.x;
    const int tid = threadIdx.x;
    const float* xr = x + (size_t)row * CC;

    float s = 0.f;
    for (int c = tid; c < CC; c += 256) s += xr[c];
    red[tid] = s; __syncthreads();
    for (int st = 128; st > 0; st >>= 1) {
        if (tid < st) red[tid] += red[tid + st];
        __syncthreads();
    }
    const float mu = red[0] * (1.0f / CC);
    __syncthreads();

    float v = 0.f;
    for (int c = tid; c < CC; c += 256) { float d = xr[c] - mu; v += d * d; }
    red[tid] = v; __syncthreads();
    for (int st = 128; st > 0; st >>= 1) {
        if (tid < st) red[tid] += red[tid + st];
        __syncthreads();
    }
    const float rs = rsqrtf(red[0] * (1.0f / CC) + 1e-5f);

    unsigned short* orow = out + (size_t)row * CC;
    for (int c = tid; c < CC; c += 256)
        orow[c] = f32_to_bf16((xr[c] - mu) * rs * g[c] + be[c]);
}

// ---------------- tiled bf16 WMMA GEMM (async-pipelined) ----------------
// C[M,Nn] = epilogue( A[Mpad,K](bf16) @ B[K,Nn](bf16) + bias )
// Block tile 64x128, K-step 64, 8 waves (2x4), 2x2 16x16 frags per wave.
// Async DMA for k-step t+1 is issued before the WMMAs of k-step t.
// MODE 0: bias   MODE 1: bias + exact GELU   MODE 2: bias + residual
template <int MODE>
__global__ __launch_bounds__(256) void gemm_bf16_kernel(
    const unsigned short* __restrict__ A, const unsigned short* __restrict__ Bw,
    const float* __restrict__ bias, const float* __restrict__ resid,
    float* __restrict__ outF, unsigned short* __restrict__ outH,
    int M, int K, int Nn) {
    __shared__ unsigned short As[64][72];     // [m][k],  144B row stride (16B aligned)
    __shared__ unsigned short Bs[64][136];    // [k][n],  272B row stride (16B aligned)

    const int tid  = threadIdx.x;
    const int lane = tid & 31;
    const int wave = tid >> 5;
    const int hi   = lane >> 4;
    const int l16  = lane & 15;
    const int wm   = wave >> 2;            // 0..1  (M waves)
    const int wn   = wave & 3;             // 0..3  (N waves)
    const int m0   = blockIdx.y * 64;
    const int n0   = blockIdx.x * 128;

    // per-thread staging coordinates (16B chunks)
    const int arow = tid >> 3, asub = tid & 7;     // A: 64 rows x 8 chunks (x2)
    const int brow = tid >> 4, bsub = tid & 15;    // B: 64 rows x 16 chunks (x4)

    auto stage = [&](int k0) {
#if HAS_ASYNC_LDS
#pragma unroll
        for (int i = 0; i < 2; ++i)
            async_cp16(A + (size_t)(m0 + arow + i * 32) * K + k0 + asub * 8,
                       &As[arow + i * 32][asub * 8]);
#pragma unroll
        for (int i = 0; i < 4; ++i)
            async_cp16(Bw + (size_t)(k0 + brow + i * 16) * Nn + n0 + bsub * 8,
                       &Bs[brow + i * 16][bsub * 8]);
#else
#pragma unroll
        for (int i = 0; i < 2; ++i)
            *(u32x4*)&As[arow + i * 32][asub * 8] =
                *(const u32x4*)(A + (size_t)(m0 + arow + i * 32) * K + k0 + asub * 8);
#pragma unroll
        for (int i = 0; i < 4; ++i)
            *(u32x4*)&Bs[brow + i * 16][bsub * 8] =
                *(const u32x4*)(Bw + (size_t)(k0 + brow + i * 16) * Nn + n0 + bsub * 8);
#endif
    };

    v8f acc[2][2];
#pragma unroll
    for (int a = 0; a < 2; ++a)
#pragma unroll
        for (int b = 0; b < 2; ++b)
#pragma unroll
            for (int r = 0; r < 8; ++r) acc[a][b][r] = 0.f;

    stage(0);                                   // prologue DMA

    for (int k0 = 0; k0 < K; k0 += 64) {
#if HAS_ASYNC_LDS
        async_wait0();                          // own copies landed
#endif
        __syncthreads();                        // => whole tile landed

#pragma unroll
        for (int c = 0; c < 2; ++c) {
            unsigned int afr[2][8], bfr[2][8];
#pragma unroll
            for (int mf = 0; mf < 2; ++mf) {
                int row = wm * 32 + mf * 16 + l16;
#pragma unroll
                for (int i = 0; i < 8; ++i)
                    afr[mf][i] =
                        *(const unsigned int*)&As[row][c * 32 + kA_idx(i, hi)];
            }
#pragma unroll
            for (int nf = 0; nf < 2; ++nf) {
                int col = wn * 32 + nf * 16 + l16;
#pragma unroll
                for (int i = 0; i < 8; ++i) {
                    int k = c * 32 + 2 * i + hi * 16;
                    unsigned int lo = Bs[k][col];
                    unsigned int hb = Bs[k + 1][col];
                    bfr[nf][i] = lo | (hb << 16);
                }
            }
            if (c == 1) {
                wait_ds0();                     // all our LDS reads are in regs
                __syncthreads();                // everyone done reading the tile
                if (k0 + 64 < K) stage(k0 + 64);  // DMA next tile under the WMMAs
            }
#pragma unroll
            for (int mf = 0; mf < 2; ++mf)
#pragma unroll
                for (int nf = 0; nf < 2; ++nf)
                    acc[mf][nf] = wmma_bf16(afr[mf], bfr[nf], acc[mf][nf]);
        }
    }

    // epilogue (store-guarded; resid/outF only touch valid rows)
#pragma unroll
    for (int mf = 0; mf < 2; ++mf)
#pragma unroll
        for (int nf = 0; nf < 2; ++nf) {
            int gcol = n0 + wn * 32 + nf * 16 + l16;
            float bv = bias[gcol];
#pragma unroll
            for (int r = 0; r < 8; ++r) {
                int grow = m0 + wm * 32 + mf * 16 + r + hi * 8;
                if (grow < M) {
                    float v = acc[mf][nf][r] + bv;
                    if (MODE == 1)
                        v = 0.5f * v * (1.0f + erff(v * 0.70710678118f));
                    if (MODE == 2)
                        v += resid[(size_t)grow * Nn + gcol];
                    size_t o = (size_t)grow * Nn + gcol;
                    if (outF) outF[o] = v;
                    if (outH) outH[o] = f32_to_bf16(v);
                }
            }
        }
}

// ---------------- fused attention per (b, h, q-tile) ----------------
// qkv: bf16 [MPAD][2304], rows >= 9232 zeroed. Columns: s*768 + h*64 + d.
// Writes fp32 attn_weights [B,H,N,N] and bf16 ctx [MPAD][768].
__global__ __launch_bounds__(256) void attention_kernel(
    const unsigned short* __restrict__ qkv, float* __restrict__ attn,
    unsigned short* __restrict__ ctx) {
    __shared__ float          sc[16][SCOLS];   // scores (fp32)
    __shared__ unsigned short p [16][SPAD];    // softmax probs (bf16), K-padded
    __shared__ float          red[16][16];
    __shared__ float          rowmax[16], rowsum[16];

    const int tid  = threadIdx.x;
    const int lane = tid & 31;
    const int wave = tid >> 5;
    const int hi   = lane >> 4;
    const int l16  = lane & 15;
    const int qt = blockIdx.x, h = blockIdx.y, b = blockIdx.z;
    const long bn0 = (long)b * NN;
    const float scale = 0.125f;               // 1/sqrt(64)

    // Q fragments (fixed for whole block of k-tiles): 2 K-chunks of 32.
    // Rows beyond 577 read next-batch or zero-pad qkv rows -> harmless, outputs unstored.
    unsigned int qf[2][8];
    {
        const unsigned short* qp = qkv + (bn0 + qt * 16 + l16) * QKV_N + h * DD;
#pragma unroll
        for (int c = 0; c < 2; ++c)
#pragma unroll
            for (int i = 0; i < 8; ++i)
                qf[c][i] = *(const unsigned int*)(qp + c * 32 + kA_idx(i, hi));
    }

    // scores = (q @ k^T) * scale, k-tiles strided across the 8 waves
    for (int kt = wave; kt < NTILES; kt += 8) {
        v8f a;
#pragma unroll
        for (int r = 0; r < 8; ++r) a[r] = 0.f;
        const unsigned short* kp =
            qkv + (bn0 + kt * 16 + l16) * QKV_N + CC + h * DD;
#pragma unroll
        for (int c = 0; c < 2; ++c) {
            unsigned int bf[8];
#pragma unroll
            for (int i = 0; i < 8; ++i)
                bf[i] = *(const unsigned int*)(kp + c * 32 + 2 * i + hi * 16);
            a = wmma_bf16(qf[c], bf, a);
        }
        int col = kt * 16 + l16;
        bool valid = col < NN;
#pragma unroll
        for (int r = 0; r < 8; ++r)
            sc[r + hi * 8][col] = valid ? a[r] * scale : -1e30f;
    }
    __syncthreads();

    // softmax: 16 threads per row
    {
        const int row = tid >> 4, j = tid & 15;
        float mx = -1e30f;
        for (int c = j; c < SCOLS; c += 16) mx = fmaxf(mx, sc[row][c]);
        red[row][j] = mx; __syncthreads();
        if (j == 0) {
            float m = -1e30f;
#pragma unroll
            for (int t = 0; t < 16; ++t) m = fmaxf(m, red[row][t]);
            rowmax[row] = m;
        }
        __syncthreads();
        const float mr = rowmax[row];
        float s = 0.f;
        for (int c = j; c < SCOLS; c += 16) s += expf(sc[row][c] - mr);
        red[row][j] = s; __syncthreads();
        if (j == 0) {
            float t2 = 0.f;
#pragma unroll
            for (int t = 0; t < 16; ++t) t2 += red[row][t];
            rowsum[row] = t2;
        }
        __syncthreads();
        const float inv = 1.0f / rowsum[row];
        const int qrow = qt * 16 + row;
        const bool rok = qrow < NN;
        float* aout = attn + (((long)(b * HH + h) * NN + qrow) * NN);
        for (int c = j; c < SPAD; c += 16) {
            if (c < SCOLS) {
                float pv = expf(sc[row][c] - mr) * inv;
                p[row][c] = f32_to_bf16(pv);
                if (rok && c < NN) aout[c] = pv;
            } else {
                p[row][c] = 0;
            }
        }
    }
    __syncthreads();

    // ctx = P @ V : waves 0..3 each produce one 16-wide slice of D=64.
    // V rows >= 577 are next-batch or zero-pad rows and p==0 there anyway.
    if (wave < 4) {
        const int nf = wave;
        v8f a;
#pragma unroll
        for (int r = 0; r < 8; ++r) a[r] = 0.f;
        const unsigned short* vp = qkv + bn0 * QKV_N + 2 * CC + h * DD + nf * 16 + l16;
        for (int s = 0; s < SPAD / 32; ++s) {
            int k0 = s * 32;
            unsigned int pa[8], vb[8];
#pragma unroll
            for (int i = 0; i < 8; ++i)
                pa[i] = *(const unsigned int*)&p[l16][k0 + kA_idx(i, hi)];
#pragma unroll
            for (int i = 0; i < 8; ++i) {
                int k = k0 + 2 * i + hi * 16;
                unsigned int lo = vp[(size_t)k * QKV_N];
                unsigned int hb = vp[(size_t)(k + 1) * QKV_N];
                vb[i] = lo | (hb << 16);
            }
            a = wmma_bf16(pa, vb, a);
        }
#pragma unroll
        for (int r = 0; r < 8; ++r) {
            int gr = qt * 16 + r + hi * 8;
            if (gr < NN)
                ctx[(bn0 + gr) * CC + h * DD + nf * 16 + l16] =
                    f32_to_bf16(a[r]);
        }
    }
}

// ---------------- host-side launch ----------------
extern "C" void kernel_launch(void* const* d_in, const int* in_sizes, int n_in,
                              void* d_out, int out_size, void* d_ws, size_t ws_size,
                              hipStream_t stream) {
    (void)in_sizes; (void)n_in; (void)out_size; (void)ws_size;
    const float* x      = (const float*)d_in[0];
    const float* w_qkv  = (const float*)d_in[1];
    const float* b_qkv  = (const float*)d_in[2];
    const float* w_proj = (const float*)d_in[3];
    const float* b_proj = (const float*)d_in[4];
    const float* g1     = (const float*)d_in[5];
    const float* be1    = (const float*)d_in[6];
    const float* g2     = (const float*)d_in[7];
    const float* be2    = (const float*)d_in[8];
    const float* w1     = (const float*)d_in[9];
    const float* b1     = (const float*)d_in[10];
    const float* w2     = (const float*)d_in[11];
    const float* b2     = (const float*)d_in[12];

    float* xout = (float*)d_out;                                   // [B,N,C]
    float* attn = xout + (size_t)MROWS * CC;                       // [B,H,N,N]

    // workspace carve-up (bytes); activations padded to MPAD rows
    char* ws = (char*)d_ws;
    size_t off = 0;
    auto carve = [&](size_t bytes) { char* q = ws + off; off += bytes; return q; };
    unsigned short* wqb  = (unsigned short*)carve((size_t)CC * QKV_N * 2);
    unsigned short* wpb  = (unsigned short*)carve((size_t)CC * CC * 2);
    unsigned short* w1b  = (unsigned short*)carve((size_t)CC * HID * 2);
    unsigned short* w2b  = (unsigned short*)carve((size_t)HID * CC * 2);
    unsigned short* xn1  = (unsigned short*)carve((size_t)MPAD * CC * 2);
    unsigned short* qkvb = (unsigned short*)carve((size_t)MPAD * QKV_N * 2);
    unsigned short* ctxb = (unsigned short*)carve((size_t)MPAD * CC * 2);
    unsigned short* xn2  = (unsigned short*)carve((size_t)MPAD * CC * 2);
    unsigned short* hb   = (unsigned short*)carve((size_t)MPAD * HID * 2);
    float*          x1   = (float*)carve((size_t)MROWS * CC * 4);

    auto cvt = [&](const float* src, unsigned short* dst, int n) {
        cvt_bf16_kernel<<<(n + 255) / 256, 256, 0, stream>>>(src, dst, n);
    };
    cvt(w_qkv, wqb, CC * QKV_N);
    cvt(w_proj, wpb, CC * CC);
    cvt(w1, w1b, CC * HID);
    cvt(w2, w2b, HID * CC);

    // zero the qkv padding rows (read unconditionally by attention)
    {
        const int nslack = (MPAD - MROWS) * QKV_N;
        zero_u16_kernel<<<(nslack + 255) / 256, 256, 0, stream>>>(
            qkvb + (size_t)MROWS * QKV_N, nslack);
    }

    // LN1
    layernorm_bf16_kernel<<<MROWS, 256, 0, stream>>>(x, g1, be1, xn1);

    const int MT = MPAD / 64;   // 145
    // QKV GEMM -> bf16
    gemm_bf16_kernel<0><<<dim3(QKV_N / 128, MT), 256, 0, stream>>>(
        xn1, wqb, b_qkv, nullptr, nullptr, qkvb, MROWS, CC, QKV_N);
    // attention -> attn_weights (fp32) + ctx (bf16)
    attention_kernel<<<dim3(NTILES, HH, BB), 256, 0, stream>>>(qkvb, attn, ctxb);
    // proj + residual(x) -> x1 (fp32)
    gemm_bf16_kernel<2><<<dim3(CC / 128, MT), 256, 0, stream>>>(
        ctxb, wpb, b_proj, x, x1, nullptr, MROWS, CC, CC);
    // LN2
    layernorm_bf16_kernel<<<MROWS, 256, 0, stream>>>(x1, g2, be2, xn2);
    // fc1 + exact GELU -> bf16
    gemm_bf16_kernel<1><<<dim3(HID / 128, MT), 256, 0, stream>>>(
        xn2, w1b, b1, nullptr, nullptr, hb, MROWS, CC, HID);
    // fc2 + residual(x1) -> d_out
    gemm_bf16_kernel<2><<<dim3(CC / 128, MT), 256, 0, stream>>>(
        hb, w2b, b2, x1, xout, nullptr, MROWS, HID, CC);
}
